// CausalSelfAttention_18743237280277
// MI455X (gfx1250) — matrix operand order repaired and checked by reading, not compiled, not run on hardware
//
#include <hip/hip_runtime.h>

// ---------------------------------------------------------------------------
// CausalSelfAttention for MI455X (gfx1250): bf16 WMMA + TDM/async-LDS staging.
// B=4, T=2048, C=768, H=12, hs=64.
// ---------------------------------------------------------------------------

#define BB 4
#define TT 2048
#define CC 768
#define HH 12
#define HSZ 64
#define MROWS (BB * TT)      // 8192
#define N3C (3 * CC)         // 2304

typedef __attribute__((ext_vector_type(16))) __bf16 v16bf;
typedef __attribute__((ext_vector_type(8)))  float  v8f;
typedef __attribute__((ext_vector_type(4)))  __bf16 v4bf;
typedef __attribute__((ext_vector_type(4)))  unsigned int v4u;
typedef __attribute__((ext_vector_type(8)))  int v8i;
typedef __attribute__((ext_vector_type(4)))  int v4i;
typedef int gv4i __attribute__((vector_size(16)));   // GCC-style int32x4 (builtin param type)

#define AS1 __attribute__((address_space(1)))
#define AS3 __attribute__((address_space(3)))

// ---- feature probes (compile-safe: fall back to sync paths) ----------------
#if defined(__has_builtin)
#if __has_builtin(__builtin_amdgcn_global_load_async_to_lds_b128)
#define HAVE_ASYNC 1
#endif
#if __has_builtin(__builtin_amdgcn_tensor_load_to_lds)
#define HAVE_TDM 1
#endif
#endif
#ifndef HAVE_ASYNC
#define HAVE_ASYNC 0
#endif
#ifndef HAVE_TDM
#define HAVE_TDM 0
#endif

__device__ __forceinline__ void wait_async0() {
#if __has_builtin(__builtin_amdgcn_s_wait_asynccnt)
  __builtin_amdgcn_s_wait_asynccnt(0);
#else
  asm volatile("s_wait_asynccnt 0" ::: "memory");
#endif
}

__device__ __forceinline__ void wait_tensor0() {
#if __has_builtin(__builtin_amdgcn_s_wait_tensorcnt)
  __builtin_amdgcn_s_wait_tensorcnt(0);
#else
  asm volatile("s_wait_tensorcnt 0" ::: "memory");
#endif
}

union FragBF {
  v16bf v;
  uint4 u[2];
};

// ---------------------------------------------------------------------------
// float32 -> bf16 conversion, 4 elements / thread
// ---------------------------------------------------------------------------
__global__ __launch_bounds__(256)
void cvt_f32_to_bf16(const float* __restrict__ src, __bf16* __restrict__ dst, int n4) {
  int i = blockIdx.x * blockDim.x + threadIdx.x;
  if (i >= n4) return;
  float4 f = ((const float4*)src)[i];
  v4bf o;
  o[0] = (__bf16)f.x; o[1] = (__bf16)f.y; o[2] = (__bf16)f.z; o[3] = (__bf16)f.w;
  ((v4bf*)dst)[i] = o;
}

// ---------------------------------------------------------------------------
// Tiled bf16 GEMM with f32 accumulation via v_wmma_f32_16x16x32_bf16.
//   WG: 256 thr = 8 waves. WG tile 64(M) x 128(N), K step 32.
//   Wave (wm,wn): 16(M) x 64(N) -> 4 accumulators of 16x16.
// mode 0: outF[m*N+n] = acc + bias[n]                   (f32)
// mode 1: scatter into q/k/v bf16 [B,H,T,hs], N == 3C
// ---------------------------------------------------------------------------
__global__ __launch_bounds__(256)
void gemm_bf16_wmma(const __bf16* __restrict__ A, const __bf16* __restrict__ Bm,
                    const float* __restrict__ bias,
                    float* __restrict__ outF,
                    __bf16* __restrict__ qd, __bf16* __restrict__ kd, __bf16* __restrict__ vd,
                    int M, int N, int K, int mode) {
  __shared__ __bf16 As[64 * 32];    // [m][k] row-major, stride 32
  __shared__ __bf16 BsT[128 * 32];  // [n][k] row-major, stride 32 (transposed tile)

  const int tid  = threadIdx.x;
  const int lane = tid & 31;
  const int wid  = tid >> 5;
  const int wm   = wid & 3;     // 0..3 -> M sub-tile
  const int wn   = wid >> 2;    // 0..1 -> N sub-tile (64 wide)
  const int half = lane >> 4;   // lane group for WMMA layouts
  const int l16  = lane & 15;

  const int nTilesN = N / 128;
  const int m0 = (blockIdx.x / nTilesN) * 64;
  const int n0 = (blockIdx.x % nTilesN) * 128;

  const v8f zero8 = {0.f, 0.f, 0.f, 0.f, 0.f, 0.f, 0.f, 0.f};
  v8f acc[4];
#pragma unroll
  for (int i = 0; i < 4; ++i) acc[i] = zero8;

  const int arow = tid >> 2;
  const int acol = (tid & 3) * 8;

  for (int k0 = 0; k0 < K; k0 += 32) {
    __syncthreads();
    // prefetch next A tile rows into L2 while we stage this one
    if (k0 + 32 < K)
      __builtin_prefetch(&A[(size_t)(m0 + arow) * K + k0 + 32 + acol], 0, 1);

    // stage A: 64x32 bf16, one b128 per thread (async to LDS when available)
#if HAVE_ASYNC
    __builtin_amdgcn_global_load_async_to_lds_b128(
        (AS1 gv4i*)&A[(size_t)(m0 + arow) * K + k0 + acol],
        (AS3 gv4i*)&As[arow * 32 + acol], 0, 0);
#else
    *(uint4*)&As[arow * 32 + acol] =
        *(const uint4*)&A[(size_t)(m0 + arow) * K + k0 + acol];
#endif
    // stage B transposed: 32x128 -> BsT[n][k]
#pragma unroll
    for (int it = 0; it < 2; ++it) {
      int krow = (tid >> 4) + 16 * it;
      int ncol = (tid & 15) * 8;
      union { uint4 u; __bf16 e[8]; } t;
      t.u = *(const uint4*)&Bm[(size_t)(k0 + krow) * N + n0 + ncol];
#pragma unroll
      for (int j = 0; j < 8; ++j) BsT[(ncol + j) * 32 + krow] = t.e[j];
    }
#if HAVE_ASYNC
    wait_async0();
#endif
    __syncthreads();

    // A fragment: row = l16, k = 16*half + e (contiguous)
    FragBF fa;
    {
      const __bf16* base = &As[(wm * 16 + l16) * 32];
      fa.u[0] = *(const uint4*)&base[16 * half];
      fa.u[1] = *(const uint4*)&base[16 * half + 8];
    }
#pragma unroll
    for (int i = 0; i < 4; ++i) {
      // B fragment: n = l16, k = 16*half + 0..15 (contiguous in BsT row)
      FragBF fb;
      const __bf16* base = &BsT[(wn * 64 + i * 16 + l16) * 32 + 16 * half];
      fb.u[0] = *(const uint4*)&base[0];
      fb.u[1] = *(const uint4*)&base[8];
      acc[i] = __builtin_amdgcn_wmma_f32_16x16x32_bf16(
          false, fa.v, false, fb.v, (short)0, acc[i], false, false);
    }
  }

  // epilogue: D layout -> lane holds n = l16, rows m = r + 8*half
#pragma unroll
  for (int i = 0; i < 4; ++i) {
    int n_g = n0 + wn * 64 + i * 16 + l16;
    float bv = bias[n_g];
#pragma unroll
    for (int r = 0; r < 8; ++r) {
      int m_g = m0 + wm * 16 + r + 8 * half;
      float val = acc[i][r] + bv;
      if (mode == 0) {
        outF[(size_t)m_g * N + n_g] = val;
      } else {
        int which = n_g / CC;
        int c = n_g - which * CC;
        int head = c >> 6;
        int d = c & (HSZ - 1);
        int b = m_g >> 11;        // / TT
        int t = m_g & (TT - 1);
        __bf16* dst = (which == 0) ? qd : (which == 1) ? kd : vd;
        dst[((((size_t)b * HH + head) * TT) + t) * HSZ + d] = (__bf16)val;
      }
    }
  }
}

// ---------------------------------------------------------------------------
// Flash attention (causal, online softmax), bf16 WMMA.
//   WG: 128 thr = 4 waves; WG owns 64 q rows (wave: 16 rows).
//   Loop over 64-wide kpos blocks (bounded by causality).
//   K tile staged by the Tensor Data Mover (wave 0 issues, TENSORcnt sync);
//   V tile staged transposed by VALU; Q held in registers.
// ---------------------------------------------------------------------------
__global__ __launch_bounds__(128)
void flash_attn_wmma(const __bf16* __restrict__ Q, const __bf16* __restrict__ Kx,
                     const __bf16* __restrict__ Vx, __bf16* __restrict__ Y) {
  __shared__ __bf16 Ks[64 * 64];        // [kpos][d]
  __shared__ __bf16 VsT[64 * 64];       // [d][kpos]
  __shared__ __bf16 Ps[4 * 16 * 64];    // per-wave P tile [q][kpos]

  const int tid  = threadIdx.x;
  const int lane = tid & 31;
  const int wid  = tid >> 5;    // 0..3
  const int half = lane >> 4;
  const int l16  = lane & 15;

  const int qt = blockIdx.x & 31;       // T/64 = 32 q tiles
  const int bh = blockIdx.x >> 5;       // 0..B*H-1
  const int h  = bh % HH;
  const int b  = bh / HH;

  const size_t base = ((size_t)b * HH + h) * TT * HSZ;
  const __bf16* qp = Q + base;
  const __bf16* kp = Kx + base;
  const __bf16* vp = Vx + base;

  const int qrow0 = qt * 64 + wid * 16;

  // Q fragments (d split into two 32-wide WMMA k-steps), loaded once
  FragBF fq[2];
  {
    const __bf16* qr = qp + (size_t)(qrow0 + l16) * HSZ;
#pragma unroll
    for (int s = 0; s < 2; ++s) {
      fq[s].u[0] = *(const uint4*)&qr[32 * s + 16 * half];
      fq[s].u[1] = *(const uint4*)&qr[32 * s + 16 * half + 8];
    }
  }

  const v8f zero8 = {0.f, 0.f, 0.f, 0.f, 0.f, 0.f, 0.f, 0.f};
  float mrow[8], lrow[8];
  v8f oacc[4];
#pragma unroll
  for (int r = 0; r < 8; ++r) { mrow[r] = -3.0e38f; lrow[r] = 0.f; }
#pragma unroll
  for (int dt = 0; dt < 4; ++dt) oacc[dt] = zero8;

#if HAVE_TDM
  const unsigned ksLds = (unsigned)(uintptr_t)&Ks[0];
#endif

  for (int kb = 0; kb <= qt; ++kb) {
    const int k0 = kb * 64;
    __syncthreads();  // previous iteration's Ks/VsT readers done

    // prefetch next K/V block into L2 (spread across the 128 threads)
    if (kb < qt) {
      __builtin_prefetch(&kp[(size_t)(k0 + 64) * HSZ + tid * 32], 0, 1);
      __builtin_prefetch(&vp[(size_t)(k0 + 64) * HSZ + tid * 32], 0, 1);
    }

#if HAVE_TDM
    // ---- K tile via Tensor Data Mover: 2D tile 16(x8B) x 64 rows ----
    if (wid == 0) {
      unsigned long long ga = (unsigned long long)(uintptr_t)(kp + (size_t)k0 * HSZ);
      v4u g0;
      g0[0] = 1u;                                   // count=1, user D#
      g0[1] = ksLds;                                // lds_addr (bytes)
      g0[2] = (unsigned)ga;                         // global_addr[31:0]
      g0[3] = (unsigned)((ga >> 32) & 0x01FFFFFFu)  // global_addr[56:32]
              | (2u << 30);                         // type = 2 (image)
      v8i g1;
      g1[0] = (int)(3u << 16);   // data_size=3 (8B), mask=0
      g1[1] = (int)(16u << 16);  // tensor_dim0 = 16 (8B units)
      g1[2] = (int)(64u << 16);  // tensor_dim1 = 64
      g1[3] = (int)(16u << 16);  // tile_dim0 = 16
      g1[4] = 64;                // tile_dim1 = 64
      g1[5] = 16;                // tensor_dim0_stride = 16
      g1[6] = 0;
      g1[7] = 0;
      v4i z4 = {0, 0, 0, 0};
#if defined(__clang_major__) && (__clang_major__ >= 23)
      v8i z8 = {0, 0, 0, 0, 0, 0, 0, 0};
      __builtin_amdgcn_tensor_load_to_lds(g0, g1, z4, z4, z8, 0);
#else
      __builtin_amdgcn_tensor_load_to_lds(g0, g1, z4, z4, 0);
#endif
    }
#endif

    // ---- V tile transposed (and K tile when no TDM): 64x64 each ----
#pragma unroll
    for (int c = 0; c < 4; ++c) {
      int chunk = tid + 128 * c;       // 512 chunks of 8 elems
      int row = chunk >> 3;
      int col = (chunk & 7) * 8;
#if !HAVE_TDM
      *(uint4*)&Ks[row * 64 + col] =
          *(const uint4*)&kp[(size_t)(k0 + row) * HSZ + col];
#endif
      union { uint4 u; __bf16 e[8]; } t;
      t.u = *(const uint4*)&vp[(size_t)(k0 + row) * HSZ + col];
#pragma unroll
      for (int j = 0; j < 8; ++j) VsT[(col + j) * 64 + row] = t.e[j];
    }
#if HAVE_TDM
    if (wid == 0) wait_tensor0();
#endif
    __syncthreads();

    // ---- scores S = Q K^T (4 tiles of 16 kpos) ----
    float sv[4][8];
#pragma unroll
    for (int nt = 0; nt < 4; ++nt) {
      v8f sacc = zero8;
#pragma unroll
      for (int s = 0; s < 2; ++s) {
        FragBF fk;  // B frag: n = kpos row l16, k = d contiguous
        const __bf16* bk = &Ks[(nt * 16 + l16) * 64 + 32 * s + 16 * half];
        fk.u[0] = *(const uint4*)&bk[0];
        fk.u[1] = *(const uint4*)&bk[8];
        sacc = __builtin_amdgcn_wmma_f32_16x16x32_bf16(
            false, fq[s].v, false, fk.v, (short)0, sacc, false, false);
      }
      int n_g = k0 + nt * 16 + l16;
#pragma unroll
      for (int r = 0; r < 8; ++r) {
        int m_g = qrow0 + r + 8 * half;
        float vvv = sacc[r] * 0.125f;  // 1/sqrt(64)
        sv[nt][r] = (n_g > m_g) ? -3.0e38f : vvv;
      }
    }

    // ---- online softmax (row = r + 8*half; 16-lane reductions) ----
    float alpha[8];
#pragma unroll
    for (int r = 0; r < 8; ++r) {
      float mx = fmaxf(fmaxf(sv[0][r], sv[1][r]), fmaxf(sv[2][r], sv[3][r]));
#pragma unroll
      for (int off = 1; off < 16; off <<= 1)
        mx = fmaxf(mx, __shfl_xor(mx, off, 32));
      float mnew = fmaxf(mrow[r], mx);
      alpha[r] = __expf(mrow[r] - mnew);
      mrow[r] = mnew;
      float rs = 0.f;
#pragma unroll
      for (int nt = 0; nt < 4; ++nt) {
        sv[nt][r] = __expf(sv[nt][r] - mnew);
        rs += sv[nt][r];
      }
#pragma unroll
      for (int off = 1; off < 16; off <<= 1)
        rs += __shfl_xor(rs, off, 32);
      lrow[r] = lrow[r] * alpha[r] + rs;
    }
#pragma unroll
    for (int dt = 0; dt < 4; ++dt)
#pragma unroll
      for (int r = 0; r < 8; ++r) oacc[dt][r] *= alpha[r];

    // ---- transpose P (D layout) -> A layout via per-wave LDS tile ----
    __bf16* pw = &Ps[wid * 16 * 64];
#pragma unroll
    for (int nt = 0; nt < 4; ++nt)
#pragma unroll
      for (int r = 0; r < 8; ++r)
        pw[(r + 8 * half) * 64 + nt * 16 + l16] = (__bf16)sv[nt][r];
    __syncthreads();

    // ---- O += P V ----
#pragma unroll
    for (int s = 0; s < 2; ++s) {
      FragBF fp;  // A frag from Ps: row = l16
      const __bf16* bp = &pw[l16 * 64 + 32 * s];
      fp.u[0] = *(const uint4*)&bp[16 * half];
      fp.u[1] = *(const uint4*)&bp[16 * half + 8];
#pragma unroll
      for (int dt = 0; dt < 4; ++dt) {
        FragBF fv;  // B frag: n = d row in VsT, k = kpos contiguous
        const __bf16* bv = &VsT[(dt * 16 + l16) * 64 + 32 * s + 16 * half];
        fv.u[0] = *(const uint4*)&bv[0];
        fv.u[1] = *(const uint4*)&bv[8];
        oacc[dt] = __builtin_amdgcn_wmma_f32_16x16x32_bf16(
            false, fp.v, false, fv.v, (short)0, oacc[dt], false, false);
      }
    }
  }

  // ---- normalize and write Y [B,T,H*hs] bf16 ----
#pragma unroll
  for (int dt = 0; dt < 4; ++dt) {
    int d = dt * 16 + l16;
#pragma unroll
    for (int r = 0; r < 8; ++r) {
      int t = qrow0 + r + 8 * half;
      float val = oacc[dt][r] / lrow[r];
      Y[(((size_t)b * TT + t) * HH + h) * HSZ + d] = (__bf16)val;
    }
  }
}

// ---------------------------------------------------------------------------
// Host-side orchestration
// ---------------------------------------------------------------------------
extern "C" void kernel_launch(void* const* d_in, const int* in_sizes, int n_in,
                              void* d_out, int out_size, void* d_ws, size_t ws_size,
                              hipStream_t stream) {
  const float* x      = (const float*)d_in[0];
  const float* w_attn = (const float*)d_in[1];
  const float* b_attn = (const float*)d_in[2];
  const float* w_proj = (const float*)d_in[3];
  const float* b_proj = (const float*)d_in[4];
  float* out = (float*)d_out;

  char* ws = (char*)d_ws;
  const size_t xbN  = (size_t)MROWS * CC;        // 6,291,456
  const size_t wabN = (size_t)CC * N3C;          // 1,769,472
  const size_t wpbN = (size_t)CC * CC;           //   589,824
  __bf16* xb  = (__bf16*)(ws);
  __bf16* wab = (__bf16*)(ws + 2 * xbN);
  __bf16* wpb = (__bf16*)(ws + 2 * (xbN + wabN));
  __bf16* qb  = (__bf16*)(ws + 2 * (xbN + wabN + wpbN));
  __bf16* kb  = qb + xbN;
  __bf16* vb  = kb + xbN;
  __bf16* yb  = vb + xbN;

  // 1) f32 -> bf16 converts
  {
    int n4 = (int)(xbN / 4);
    cvt_f32_to_bf16<<<(n4 + 255) / 256, 256, 0, stream>>>(x, xb, n4);
    n4 = (int)(wabN / 4);
    cvt_f32_to_bf16<<<(n4 + 255) / 256, 256, 0, stream>>>(w_attn, wab, n4);
    n4 = (int)(wpbN / 4);
    cvt_f32_to_bf16<<<(n4 + 255) / 256, 256, 0, stream>>>(w_proj, wpb, n4);
  }

  // 2) QKV projection: [8192 x 768] @ [768 x 2304] -> q/k/v bf16 [B,H,T,hs]
  {
    int grid = (MROWS / 64) * (N3C / 128);  // 128 * 18 = 2304
    gemm_bf16_wmma<<<grid, 256, 0, stream>>>(xb, wab, b_attn, nullptr,
                                             qb, kb, vb, MROWS, N3C, CC, 1);
  }

  // 3) Causal flash attention -> yb bf16 [B,T,C]
  {
    int grid = BB * HH * (TT / 64);  // 1536
    flash_attn_wmma<<<grid, 128, 0, stream>>>(qb, kb, vb, yb);
  }

  // 4) Output projection: [8192 x 768] @ [768 x 768] + bias -> f32 out
  {
    int grid = (MROWS / 64) * (CC / 128);  // 128 * 6 = 768
    gemm_bf16_wmma<<<grid, 256, 0, stream>>>(yb, wpb, b_proj, out,
                                             nullptr, nullptr, nullptr,
                                             MROWS, CC, CC, 0);
  }
}
